// MambaEncoder_24644522345227
// MI455X (gfx1250) — compile-verified, matrix operand order checked by reading, not run on hardware
//
#include <hip/hip_runtime.h>
#include <hip/hip_bf16.h>

#define D_MODEL 1024
#define D_INNER 2048
#define DT_RANK 64
#define D_STATE 16
#define NLAYERS 4
#define BATCH   2
#define SEQLEN  4096
#define NTOK    (BATCH*SEQLEN)      /* 8192 */
#define XPROJ_N (DT_RANK + 2*D_STATE) /* 96 */
#define CHUNK   64
#define NCHUNK  (SEQLEN/CHUNK)      /* 64 */
#define NCH     (BATCH*D_INNER)     /* 4096 */

typedef unsigned short u16b;                                      /* bf16 bits */
typedef __attribute__((ext_vector_type(4)))  unsigned int uint4v; /* 16B chunk */
typedef __attribute__((ext_vector_type(16))) __bf16 v16bf;
typedef __attribute__((ext_vector_type(8)))  float  v8f;

__device__ __forceinline__ u16b f2bf(float f) {
  unsigned u = __float_as_uint(f);
  u += 0x7FFFu + ((u >> 16) & 1u);     /* round-to-nearest-even */
  return (u16b)(u >> 16);
}
__device__ __forceinline__ float bf2f(u16b h) {
  return __uint_as_float(((unsigned)h) << 16);
}
__device__ __forceinline__ float siluf(float x) { return x / (1.f + __expf(-x)); }

/* gfx1250 async copy: global -> LDS (ASYNCcnt), bypasses VGPR staging */
__device__ __forceinline__ void async_copy_b128(unsigned lds_addr, const void* gaddr) {
  asm volatile("global_load_async_to_lds_b128 %0, %1, off"
               :: "v"(lds_addr), "v"((unsigned long long)(size_t)gaddr)
               : "memory");
}
__device__ __forceinline__ void wait_async0() {
  asm volatile("s_wait_asynccnt 0x0" ::: "memory");
}

/* ------------------------------------------------------------------ */
/* Generic bf16 GEMM:  C[M=8192, N] = A[M,K] * W[N,K]^T  (+bias,+Cin) */
/* 256 thr (8 waves); tile 128x64; BK=32; double-buffered async-LDS   */
/* pipeline feeding v_wmma_f32_16x16x32_bf16                          */
/* ------------------------------------------------------------------ */
union ABu { uint4v q[2]; v16bf v; };

__global__ __launch_bounds__(256)
void gemm_bf16_kernel(const u16b* __restrict__ A, const u16b* __restrict__ W,
                      const float* __restrict__ bias, const float* __restrict__ Cin,
                      float* __restrict__ Cf, u16b* __restrict__ Cb,
                      int N, int K, int act)
{
  __shared__ __align__(16) u16b As[2][128 * 40];   /* 80B row stride (pad) */
  __shared__ __align__(16) u16b Bs[2][64 * 40];

  const int tid   = threadIdx.x;
  const int wave  = tid >> 5;
  const int lane  = tid & 31;
  const int mBase = blockIdx.y * 128;
  const int nBase = blockIdx.x * 64;
  const int hsel  = lane >> 4;     /* 0: K 0-7/16-23 ; 1: K 8-15/24-31 */
  const int r     = lane & 15;

  /* per-thread fixed load slots */
  const int aRow0 = tid >> 2,          aQuad = tid & 3;        /* +  [0..63]  */
  const int aRow1 = (tid + 256) >> 2;                          /*    [64..127]*/
  int bRow = nBase + (tid >> 2);
  if (bRow > N - 1) bRow = N - 1;                              /* clamp edge  */
  const size_t aOff0 = (size_t)(mBase + aRow0) * K + aQuad * 8;
  const size_t aOff1 = (size_t)(mBase + aRow1) * K + aQuad * 8;
  const size_t bOff  = (size_t)bRow * K + aQuad * 8;
  const unsigned ldsA0[2] = { (unsigned)(size_t)&As[0][aRow0 * 40 + aQuad * 8],
                              (unsigned)(size_t)&As[1][aRow0 * 40 + aQuad * 8] };
  const unsigned ldsA1[2] = { (unsigned)(size_t)&As[0][aRow1 * 40 + aQuad * 8],
                              (unsigned)(size_t)&As[1][aRow1 * 40 + aQuad * 8] };
  const unsigned ldsB [2] = { (unsigned)(size_t)&Bs[0][(tid >> 2) * 40 + aQuad * 8],
                              (unsigned)(size_t)&Bs[1][(tid >> 2) * 40 + aQuad * 8] };

  v8f acc[4];
  for (int i = 0; i < 4; ++i)
    for (int j = 0; j < 8; ++j) acc[i][j] = 0.f;

  const int nk = K >> 5;
  /* prologue: stage 0 -> buffer 0 */
  async_copy_b128(ldsA0[0], A + aOff0);
  async_copy_b128(ldsA1[0], A + aOff1);
  async_copy_b128(ldsB [0], W + bOff);

  for (int i = 0; i < nk; ++i) {
    wait_async0();          /* stage i DMA (issued last iter) complete   */
    __syncthreads();        /* tile visible; prev reads of other buf done */
    if (i + 1 < nk) {       /* overlap stage i+1 DMA with stage i WMMAs  */
      int nb = (i + 1) & 1;
      size_t kh = (size_t)(i + 1) * 32;
      async_copy_b128(ldsA0[nb], A + aOff0 + kh);
      async_copy_b128(ldsA1[nb], A + aOff1 + kh);
      async_copy_b128(ldsB [nb], W + bOff  + kh);
    }
    const int buf = i & 1;
    ABu a;
    const u16b* arow = &As[buf][(wave * 16 + r) * 40];
    a.q[0] = *(const uint4v*)(arow + hsel * 8);
    a.q[1] = *(const uint4v*)(arow + 16 + hsel * 8);
#pragma unroll
    for (int n16 = 0; n16 < 4; ++n16) {
      ABu b;
      const u16b* brow = &Bs[buf][(n16 * 16 + r) * 40];
      b.q[0] = *(const uint4v*)(brow + hsel * 8);
      b.q[1] = *(const uint4v*)(brow + 16 + hsel * 8);
      acc[n16] = __builtin_amdgcn_wmma_f32_16x16x32_bf16(
          false, a.v, false, b.v, (short)0, acc[n16], false, false);
    }
  }

  /* epilogue: VGPR j -> row j (lanes 0-15) / j+8 (lanes 16-31), col = lane&15 */
#pragma unroll
  for (int n16 = 0; n16 < 4; ++n16) {
    int c = nBase + n16 * 16 + r;
    if (c >= N) continue;
    float bv = bias ? bias[c] : 0.f;
#pragma unroll
    for (int j = 0; j < 8; ++j) {
      int row = mBase + wave * 16 + j + hsel * 8;
      float v = acc[n16][j] + bv;
      if (act == 1) v = (v > 20.f) ? v : log1pf(__expf(v)); /* softplus */
      size_t o = (size_t)row * N + c;
      if (Cin) v += Cin[o];
      if (Cb) Cb[o] = f2bf(v);
      else    Cf[o] = v;
    }
  }
}

/* ------------------------------------------------------------------ */
/* LayerNorm over D_MODEL, bf16 output. One wave per token.           */
/* ------------------------------------------------------------------ */
__global__ __launch_bounds__(256)
void ln_bf16_kernel(const float* __restrict__ x, const float* __restrict__ w,
                    const float* __restrict__ bb, u16b* __restrict__ out)
{
  int wave = threadIdx.x >> 5, lane = threadIdx.x & 31;
  int token = blockIdx.x * 8 + wave;
  const float* xr = x + (size_t)token * D_MODEL;
  float s = 0.f, ss = 0.f;
#pragma unroll
  for (int i = 0; i < D_MODEL / 32; ++i) {
    float v = xr[lane + i * 32];
    s += v; ss += v * v;
  }
  for (int m = 16; m >= 1; m >>= 1) {
    s  += __shfl_xor(s, m, 32);
    ss += __shfl_xor(ss, m, 32);
  }
  float mu = s * (1.f / D_MODEL);
  float var = ss * (1.f / D_MODEL) - mu * mu;
  float rstd = rsqrtf(var + 1e-5f);
  u16b* orow = out + (size_t)token * D_MODEL;
#pragma unroll
  for (int i = 0; i < D_MODEL / 32; ++i) {
    int idx = lane + i * 32;
    orow[idx] = f2bf((xr[idx] - mu) * rstd * w[idx] + bb[idx]);
  }
}

/* mask -> zero, then final LayerNorm, f32 output */
__global__ __launch_bounds__(256)
void final_ln_kernel(const float* __restrict__ x, const unsigned char* __restrict__ mask,
                     const float* __restrict__ w, const float* __restrict__ bb,
                     float* __restrict__ out)
{
  int wave = threadIdx.x >> 5, lane = threadIdx.x & 31;
  int token = blockIdx.x * 8 + wave;
  const bool mk = mask[token] != 0;
  const float* xr = x + (size_t)token * D_MODEL;
  float s = 0.f, ss = 0.f;
#pragma unroll
  for (int i = 0; i < D_MODEL / 32; ++i) {
    float v = mk ? 0.f : xr[lane + i * 32];
    s += v; ss += v * v;
  }
  for (int m = 16; m >= 1; m >>= 1) {
    s  += __shfl_xor(s, m, 32);
    ss += __shfl_xor(ss, m, 32);
  }
  float mu = s * (1.f / D_MODEL);
  float var = ss * (1.f / D_MODEL) - mu * mu;
  float rstd = rsqrtf(var + 1e-5f);
  float* orow = out + (size_t)token * D_MODEL;
#pragma unroll
  for (int i = 0; i < D_MODEL / 32; ++i) {
    int idx = lane + i * 32;
    float v = mk ? 0.f : xr[idx];
    orow[idx] = (v - mu) * rstd * w[idx] + bb[idx];
  }
}

/* causal depthwise conv (k=4) + bias + SiLU; write f32 (scan) + bf16 (gemm A) */
__global__ __launch_bounds__(256)
void conv_silu_kernel(const u16b* __restrict__ xz, const float* __restrict__ cw,
                      const float* __restrict__ cb, float* __restrict__ uf,
                      u16b* __restrict__ ub)
{
  int idx = blockIdx.x * 256 + threadIdx.x;   /* NTOK*D_INNER */
  int d   = idx & (D_INNER - 1);
  int tok = idx >> 11;
  int t   = tok & (SEQLEN - 1);
  float acc = cb[d];
#pragma unroll
  for (int k = 0; k < 4; ++k) {
    int tt = t - 3 + k;
    if (tt >= 0)
      acc += cw[d * 4 + k] * bf2f(xz[(size_t)(tok - 3 + k) * (2 * D_INNER) + d]);
  }
  float v = siluf(acc);
  uf[idx] = v;
  ub[idx] = f2bf(v);
}

/* x_dbl[:, :64] -> bf16 A-operand for dt GEMM */
__global__ __launch_bounds__(256)
void split_dtin_kernel(const float* __restrict__ xdbl, u16b* __restrict__ dtin)
{
  int idx = blockIdx.x * 256 + threadIdx.x;   /* NTOK*DT_RANK */
  int tok = idx >> 6, c = idx & 63;
  dtin[idx] = f2bf(xdbl[(size_t)tok * XPROJ_N + c]);
}

/* ---------------- 3-phase chunked selective scan -------------------- */
/* Phase A: per (b,d,chunk): P = prod a_t, H = chunk-local state (h0=0) */
__global__ __launch_bounds__(256)
void scan_partial_kernel(const float* __restrict__ dt, const float* __restrict__ uf,
                         const float* __restrict__ xdbl, const float* __restrict__ Alog,
                         float* __restrict__ P, float* __restrict__ H)
{
  int idx = blockIdx.x * 256 + threadIdx.x;   /* NCH*NCHUNK */
  int c  = idx & (NCHUNK - 1);
  int ch = idx >> 6;
  int d  = ch & (D_INNER - 1);
  int b  = ch >> 11;
  float As[D_STATE], Pp[D_STATE], h[D_STATE];
#pragma unroll
  for (int s = 0; s < D_STATE; ++s) {
    As[s] = -__expf(Alog[d * D_STATE + s]);
    Pp[s] = 1.f; h[s] = 0.f;
  }
  int tokBase = b * SEQLEN + c * CHUNK;
  for (int tt = 0; tt < CHUNK; ++tt) {
    int tok = tokBase + tt;
    float dtv = dt[(size_t)tok * D_INNER + d];
    float du  = dtv * uf[(size_t)tok * D_INNER + d];
    const float* Bp = xdbl + (size_t)tok * XPROJ_N + DT_RANK;
#pragma unroll
    for (int s = 0; s < D_STATE; ++s) {
      float a = __expf(dtv * As[s]);
      h[s] = a * h[s] + du * Bp[s];
      Pp[s] *= a;
    }
  }
  size_t o = (size_t)idx * D_STATE;
#pragma unroll
  for (int s = 0; s < D_STATE; ++s) { P[o + s] = Pp[s]; H[o + s] = h[s]; }
}

/* Phase B: sequential prefix over 64 chunks per (b,d) -> chunk-initial states */
__global__ __launch_bounds__(256)
void scan_prefix_kernel(const float* __restrict__ P, const float* __restrict__ H,
                        float* __restrict__ Hinit)
{
  int ch = blockIdx.x * 256 + threadIdx.x;    /* NCH */
  float h[D_STATE];
#pragma unroll
  for (int s = 0; s < D_STATE; ++s) h[s] = 0.f;
  for (int c = 0; c < NCHUNK; ++c) {
    size_t o = ((size_t)ch * NCHUNK + c) * D_STATE;
#pragma unroll
    for (int s = 0; s < D_STATE; ++s) {
      Hinit[o + s] = h[s];
      h[s] = P[o + s] * h[s] + H[o + s];
    }
  }
}

/* Phase C: replay each chunk from its true initial state, emit y */
__global__ __launch_bounds__(256)
void scan_emit_kernel(const float* __restrict__ dt, const float* __restrict__ uf,
                      const float* __restrict__ xdbl, const float* __restrict__ Alog,
                      const float* __restrict__ Hinit, float* __restrict__ ys)
{
  int idx = blockIdx.x * 256 + threadIdx.x;   /* NCH*NCHUNK */
  int c  = idx & (NCHUNK - 1);
  int ch = idx >> 6;
  int d  = ch & (D_INNER - 1);
  int b  = ch >> 11;
  float As[D_STATE], h[D_STATE];
  size_t ho = (size_t)idx * D_STATE;
#pragma unroll
  for (int s = 0; s < D_STATE; ++s) {
    As[s] = -__expf(Alog[d * D_STATE + s]);
    h[s] = Hinit[ho + s];
  }
  int tokBase = b * SEQLEN + c * CHUNK;
  for (int tt = 0; tt < CHUNK; ++tt) {
    int tok = tokBase + tt;
    if (tt + 4 < CHUNK) {   /* gfx1250 global_prefetch */
      __builtin_prefetch(&dt[(size_t)(tok + 4) * D_INNER + d], 0, 1);
      __builtin_prefetch(&uf[(size_t)(tok + 4) * D_INNER + d], 0, 1);
    }
    float dtv = dt[(size_t)tok * D_INNER + d];
    float du  = dtv * uf[(size_t)tok * D_INNER + d];
    const float* Bp = xdbl + (size_t)tok * XPROJ_N + DT_RANK;
    const float* Cp = Bp + D_STATE;
    float y = 0.f;
#pragma unroll
    for (int s = 0; s < D_STATE; ++s) {
      float a = __expf(dtv * As[s]);
      h[s] = a * h[s] + du * Bp[s];
      y += h[s] * Cp[s];
    }
    ys[(size_t)tok * D_INNER + d] = y;
  }
}

/* y = (ys + u*D) * silu(z) -> bf16 (A-operand of out GEMM) */
__global__ __launch_bounds__(256)
void gate_kernel(const float* __restrict__ ys, const float* __restrict__ uf,
                 const u16b* __restrict__ xz, const float* __restrict__ Dp,
                 u16b* __restrict__ yb)
{
  int idx = blockIdx.x * 256 + threadIdx.x;   /* NTOK*D_INNER */
  int d   = idx & (D_INNER - 1);
  int tok = idx >> 11;
  float z = bf2f(xz[(size_t)tok * (2 * D_INNER) + D_INNER + d]);
  float y = (ys[idx] + uf[idx] * Dp[d]) * siluf(z);
  yb[idx] = f2bf(y);
}

/* f32 -> bf16 (weight conversion), grid-stride */
__global__ __launch_bounds__(256)
void cvt_bf16_kernel(const float* __restrict__ in, u16b* __restrict__ out, int n)
{
  int i = blockIdx.x * 256 + threadIdx.x;
  int stride = gridDim.x * 256;
  for (; i < n; i += stride) out[i] = f2bf(in[i]);
}

/* ------------------------------------------------------------------ */
extern "C" void kernel_launch(void* const* d_in, const int* in_sizes, int n_in,
                              void* d_out, int out_size, void* d_ws, size_t ws_size,
                              hipStream_t stream)
{
  (void)in_sizes; (void)n_in; (void)out_size; (void)ws_size;
  const float*         src    = (const float*)d_in[0];
  const unsigned char* mask   = (const unsigned char*)d_in[1];
  const float*         ln_w   = (const float*)d_in[2];
  const float*         ln_b   = (const float*)d_in[3];
  const float*         in_w   = (const float*)d_in[4];
  const float*         conv_w = (const float*)d_in[5];
  const float*         conv_b = (const float*)d_in[6];
  const float*         xproj_w= (const float*)d_in[7];
  const float*         dt_w   = (const float*)d_in[8];
  const float*         dt_b   = (const float*)d_in[9];
  const float*         A_log  = (const float*)d_in[10];
  const float*         Dparam = (const float*)d_in[11];
  const float*         out_w  = (const float*)d_in[12];
  const float*         fln_w  = (const float*)d_in[13];
  const float*         fln_b  = (const float*)d_in[14];

  char* p = (char*)d_ws;
  auto carve = [&](size_t bytes) -> void* {
    void* r = (void*)p;
    p += (bytes + 255) & ~(size_t)255;
    return r;
  };

  float* res   = (float*)carve((size_t)NTOK * D_MODEL * 4);
  u16b*  xn    = (u16b*) carve((size_t)NTOK * D_MODEL * 2);
  u16b*  xz    = (u16b*) carve((size_t)NTOK * 2 * D_INNER * 2);
  float* uf    = (float*)carve((size_t)NTOK * D_INNER * 4);
  u16b*  ub    = (u16b*) carve((size_t)NTOK * D_INNER * 2);
  float* xdbl  = (float*)carve((size_t)NTOK * XPROJ_N * 4);
  u16b*  dtin  = (u16b*) carve((size_t)NTOK * DT_RANK * 2);
  float* dtf   = (float*)carve((size_t)NTOK * D_INNER * 4);
  float* ysb   = (float*)carve((size_t)NTOK * D_INNER * 4);
  u16b*  yb    = (u16b*) carve((size_t)NTOK * D_INNER * 2);
  float* Pbuf  = (float*)carve((size_t)NCH * NCHUNK * D_STATE * 4);
  float* Hbuf  = (float*)carve((size_t)NCH * NCHUNK * D_STATE * 4);
  float* Hini  = (float*)carve((size_t)NCH * NCHUNK * D_STATE * 4);
  u16b*  w_in  = (u16b*) carve((size_t)NLAYERS * 2 * D_INNER * D_MODEL * 2);
  u16b*  w_xp  = (u16b*) carve((size_t)NLAYERS * XPROJ_N * D_INNER * 2);
  u16b*  w_dt  = (u16b*) carve((size_t)NLAYERS * D_INNER * DT_RANK * 2);
  u16b*  w_out = (u16b*) carve((size_t)NLAYERS * D_MODEL * D_INNER * 2);

  cvt_bf16_kernel<<<2048, 256, 0, stream>>>(in_w,   w_in,  NLAYERS * 2 * D_INNER * D_MODEL);
  cvt_bf16_kernel<<<512,  256, 0, stream>>>(xproj_w,w_xp,  NLAYERS * XPROJ_N * D_INNER);
  cvt_bf16_kernel<<<512,  256, 0, stream>>>(dt_w,   w_dt,  NLAYERS * D_INNER * DT_RANK);
  cvt_bf16_kernel<<<2048, 256, 0, stream>>>(out_w,  w_out, NLAYERS * D_MODEL * D_INNER);

  const float* cur = src;
  for (int i = 0; i < NLAYERS; ++i) {
    ln_bf16_kernel<<<NTOK / 8, 256, 0, stream>>>(cur, ln_w + i * D_MODEL,
                                                 ln_b + i * D_MODEL, xn);
    /* xz = LN(x) @ in_w^T  (bf16 out) */
    gemm_bf16_kernel<<<dim3((2 * D_INNER + 63) / 64, NTOK / 128), 256, 0, stream>>>(
        xn, w_in + (size_t)i * 2 * D_INNER * D_MODEL, nullptr, nullptr,
        nullptr, xz, 2 * D_INNER, D_MODEL, 0);
    conv_silu_kernel<<<NTOK * D_INNER / 256, 256, 0, stream>>>(
        xz, conv_w + i * D_INNER * 4, conv_b + i * D_INNER, uf, ub);
    /* x_dbl = u @ xproj_w^T */
    gemm_bf16_kernel<<<dim3((XPROJ_N + 63) / 64, NTOK / 128), 256, 0, stream>>>(
        ub, w_xp + (size_t)i * XPROJ_N * D_INNER, nullptr, nullptr,
        xdbl, nullptr, XPROJ_N, D_INNER, 0);
    split_dtin_kernel<<<NTOK * DT_RANK / 256, 256, 0, stream>>>(xdbl, dtin);
    /* dt = softplus(x_dbl[:, :64] @ dt_w^T + dt_b)  (fused bias+softplus) */
    gemm_bf16_kernel<<<dim3((D_INNER + 63) / 64, NTOK / 128), 256, 0, stream>>>(
        dtin, w_dt + (size_t)i * D_INNER * DT_RANK, dt_b + i * D_INNER, nullptr,
        dtf, nullptr, D_INNER, DT_RANK, 1);
    /* chunked selective scan */
    scan_partial_kernel<<<NCH * NCHUNK / 256, 256, 0, stream>>>(
        dtf, uf, xdbl, A_log + (size_t)i * D_INNER * D_STATE, Pbuf, Hbuf);
    scan_prefix_kernel<<<NCH / 256, 256, 0, stream>>>(Pbuf, Hbuf, Hini);
    scan_emit_kernel<<<NCH * NCHUNK / 256, 256, 0, stream>>>(
        dtf, uf, xdbl, A_log + (size_t)i * D_INNER * D_STATE, Hini, ysb);
    gate_kernel<<<NTOK * D_INNER / 256, 256, 0, stream>>>(
        ysb, uf, xz, Dparam + i * D_INNER, yb);
    /* out = residual + y @ out_w^T  (fused residual add) */
    gemm_bf16_kernel<<<dim3((D_MODEL + 63) / 64, NTOK / 128), 256, 0, stream>>>(
        yb, w_out + (size_t)i * D_MODEL * D_INNER, nullptr, cur,
        res, nullptr, D_MODEL, D_INNER, 0);
    cur = res;
  }
  final_ln_kernel<<<NTOK / 8, 256, 0, stream>>>(res, mask, fln_w, fln_b, (float*)d_out);
}
